// PathfindingGNN_17274358464713
// MI455X (gfx1250) — matrix operand order repaired and checked by reading, not compiled, not run on hardware
//
#include <hip/hip_runtime.h>
#include <hip/hip_bf16.h>

// ---------------------------------------------------------------------------
// PathfindingGNN on gfx1250 (MI455X).
//   N=100000 nodes, E=1200000 edges, F_IN=6, H=64, L=3.
//   Memory-bound: edge gather/scatter-max (~2 GB irregular traffic, L2-resident
//   at 192MB L2). GEMMs (~6 GFLOP) run on v_wmma_f32_16x16x32_f16.
// ---------------------------------------------------------------------------

typedef __attribute__((ext_vector_type(16))) _Float16 v16h;
typedef __attribute__((ext_vector_type(8)))  float    v8f;

#define NN   100000
#define EE   1200000
#define FIN  6
#define HH   64
#define LL   3
#define BNEPS 1e-5f

// ---- order-preserving float<->uint keys for atomic scatter-max -------------
__device__ __forceinline__ unsigned fkey(float f) {
    unsigned u = __float_as_uint(f);
    return (f >= 0.0f) ? (u | 0x80000000u) : ~u;
}
__device__ __forceinline__ float fdec(unsigned k) {
    float f = (k & 0x80000000u) ? __uint_as_float(k & 0x7FFFFFFFu)
                                : __uint_as_float(~k);
    // key==0 (empty segment) decodes non-finite -> reference fills with 0
    return __builtin_isfinite(f) ? f : 0.0f;
}

// ---- node encoder: h = x @ W_enc + b_enc  (K=6, VALU, memory-bound) --------
__global__ void k_encoder(const float* __restrict__ x,
                          const float* __restrict__ W,
                          const float* __restrict__ b,
                          float* __restrict__ h) {
    int t = blockIdx.x * 256 + threadIdx.x;       // NN*HH threads exactly
    int n = t >> 6, f = t & 63;
    float acc = b[f];
#pragma unroll
    for (int k = 0; k < FIN; ++k)
        acc = fmaf(x[n * FIN + k], W[k * HH + f], acc);
    h[t] = acc;
}

// ---- zero the aggregation key buffer (key 0 == "empty segment") ------------
__global__ void k_init_agg(unsigned* __restrict__ agg) {
    agg[blockIdx.x * 256 + threadIdx.x] = 0u;
}

// ---- message + scatter-max: agg[dst] = max(agg[dst], h[src]*edge_feat) -----
// 16 threads/edge, 4 features each: float4 gather, 4x global_atomic_max_u32.
__global__ void k_message(const float* __restrict__ h,
                          const int*   __restrict__ ei,   // [2,E]
                          const float* __restrict__ ea,   // [E,1]
                          const float* __restrict__ We,   // [H] (layer slice)
                          const float* __restrict__ be,   // [H]
                          unsigned*    __restrict__ agg) {
    int t = blockIdx.x * 256 + threadIdx.x;       // EE*16 threads exactly
    int e = t >> 4;
    int f = (t & 15) << 2;
    int s = ei[e];
    int d = ei[EE + e];
    float a = ea[e];
    const float4 hv = *(const float4*)(h  + s * HH + f);
    const float4 wv = *(const float4*)(We + f);
    const float4 bv = *(const float4*)(be + f);
    unsigned* dst = agg + d * HH + f;
    atomicMax(dst + 0, fkey(hv.x * fmaf(a, wv.x, bv.x)));
    atomicMax(dst + 1, fkey(hv.y * fmaf(a, wv.y, bv.y)));
    atomicMax(dst + 2, fkey(hv.z * fmaf(a, wv.z, bv.z)));
    atomicMax(dst + 3, fkey(hv.w * fmaf(a, wv.w, bv.w)));
}

// ---- layer update: hout = relu(BN(relu(cat[h,agg] @ Wu + bu))) -------------
// One 16x16 output tile per wave; 4 waves/block cover all 64 output features
// for a 16-node tile. K=128 -> 4 WMMA steps of 32 (f16 in, f32 accumulate).
__global__ void k_update(const float*    __restrict__ h,
                         const unsigned* __restrict__ agg,
                         const float*    __restrict__ Wu,   // [128,64]
                         const float*    __restrict__ bu,   // [64]
                         const float*    __restrict__ gam,
                         const float*    __restrict__ bet,
                         const float*    __restrict__ mean,
                         const float*    __restrict__ var,
                         float*          __restrict__ hout) {
    const int lane    = threadIdx.x & 31;
    const int wave    = threadIdx.x >> 5;       // 0..3 => column tile
    const int rowBase = blockIdx.x * 16;
    const int colBase = wave * 16;
    const int m       = lane & 15;
    const int hi      = lane >> 4;
    const int row     = rowBase + m;

    v8f c = {};
#pragma unroll
    for (int kk = 0; kk < 4; ++kk) {
        // A fragment (ISA 16-bit A 16x32 layout): lane<16 holds M=lane,
        // K = {base..base+7} U {base+16..base+23}; lanes 16-31 shift base by 8.
        const int kA = kk * 32 + hi * 8;
        v16h av, bv;
#pragma unroll
        for (int j = 0; j < 8; ++j) {
            int k1 = kA + j;
            int k2 = kA + 16 + j;
            float f1 = (k1 < HH) ? h[row * HH + k1] : fdec(agg[row * HH + k1 - HH]);
            float f2 = (k2 < HH) ? h[row * HH + k2] : fdec(agg[row * HH + k2 - HH]);
            av[j]     = (_Float16)f1;
            av[8 + j] = (_Float16)f2;
        }
        // B fragment: lane holds N-col = lane%16, K in the same order as A.
        const float* wcol = Wu + colBase + m;
#pragma unroll
        for (int j = 0; j < 16; ++j) bv[j] = (_Float16)wcol[(kA + j + ((j >> 3) << 3)) * HH];

        c = __builtin_amdgcn_wmma_f32_16x16x32_f16(
                false, av, false, bv, (short)0, c, false, false);
    }

    // Epilogue: bias -> relu -> BN(eval) -> relu. C/D layout: lane n-col,
    // VGPR r holds M=r (+8 for upper lane half).
    const int n    = colBase + (lane & 15);
    const int mOff = (lane >> 4) * 8;
    float bb = bu[n];
    float sc = gam[n] * rsqrtf(var[n] + BNEPS);
    float sh = bet[n] - mean[n] * sc;
#pragma unroll
    for (int r = 0; r < 8; ++r) {
        float v = c[r] + bb;
        v = fmaxf(v, 0.0f);          // relu(lin_update)
        v = fmaf(v, sc, sh);         // batch-norm (eval)
        v = fmaxf(v, 0.0f);          // final relu
        hout[(rowBase + mOff + r) * HH + n] = v;
    }
}

// ---- predictor stage 1: tmp = relu(h @ W_p1 + b_p1)  (K=64, WMMA) ----------
__global__ void k_pred1(const float* __restrict__ h,
                        const float* __restrict__ Wp,   // [64,64]
                        const float* __restrict__ bp,   // [64]
                        float*       __restrict__ tmp) {
    const int lane    = threadIdx.x & 31;
    const int wave    = threadIdx.x >> 5;
    const int rowBase = blockIdx.x * 16;
    const int colBase = wave * 16;
    const int m       = lane & 15;
    const int hi      = lane >> 4;
    const int row     = rowBase + m;

    v8f c = {};
#pragma unroll
    for (int kk = 0; kk < 2; ++kk) {
        const int kA = kk * 32 + hi * 8;
        v16h av, bv;
#pragma unroll
        for (int j = 0; j < 8; ++j) {
            av[j]     = (_Float16)h[row * HH + kA + j];
            av[8 + j] = (_Float16)h[row * HH + kA + 16 + j];
        }
        const float* wcol = Wp + colBase + m;
#pragma unroll
        for (int j = 0; j < 16; ++j) bv[j] = (_Float16)wcol[(kA + j + ((j >> 3) << 3)) * HH];
        c = __builtin_amdgcn_wmma_f32_16x16x32_f16(
                false, av, false, bv, (short)0, c, false, false);
    }
    const int n    = colBase + (lane & 15);
    const int mOff = (lane >> 4) * 8;
    float bb = bp[n];
#pragma unroll
    for (int r = 0; r < 8; ++r)
        tmp[(rowBase + mOff + r) * HH + n] = fmaxf(c[r] + bb, 0.0f);
}

// ---- predictor stage 2: out = tmp @ W_p2 + b_p2  (matvec) ------------------
__global__ void k_pred2(const float* __restrict__ tmp,
                        const float* __restrict__ w,    // [64,1]
                        const float* __restrict__ b,    // [1]
                        float*       __restrict__ out) {
    int n = blockIdx.x * 256 + threadIdx.x;
    if (n >= NN) return;
    float acc = b[0];
    const float4* tr = (const float4*)(tmp + n * HH);
    const float4* wr = (const float4*)w;
#pragma unroll
    for (int j = 0; j < 16; ++j) {
        float4 a = tr[j], ww = wr[j];
        acc = fmaf(a.x, ww.x, acc);
        acc = fmaf(a.y, ww.y, acc);
        acc = fmaf(a.z, ww.z, acc);
        acc = fmaf(a.w, ww.w, acc);
    }
    out[n] = acc;
}

// ---------------------------------------------------------------------------
extern "C" void kernel_launch(void* const* d_in, const int* in_sizes, int n_in,
                              void* d_out, int out_size, void* d_ws, size_t ws_size,
                              hipStream_t stream) {
    const float* x      = (const float*)d_in[0];
    const int*   ei     = (const int*)  d_in[1];
    const float* ea     = (const float*)d_in[2];
    const float* W_enc  = (const float*)d_in[3];
    const float* b_enc  = (const float*)d_in[4];
    const float* W_edge = (const float*)d_in[5];   // [L,1,H]
    const float* b_edge = (const float*)d_in[6];   // [L,H]
    const float* W_upd  = (const float*)d_in[7];   // [L,128,64]
    const float* b_upd  = (const float*)d_in[8];   // [L,H]
    const float* bn_g   = (const float*)d_in[9];
    const float* bn_b   = (const float*)d_in[10];
    const float* bn_m   = (const float*)d_in[11];
    const float* bn_v   = (const float*)d_in[12];
    const float* W_p1   = (const float*)d_in[13];
    const float* b_p1   = (const float*)d_in[14];
    const float* W_p2   = (const float*)d_in[15];
    const float* b_p2   = (const float*)d_in[16];
    float* out = (float*)d_out;

    // workspace carving: hA | hB | agg  (each N*H elements)
    float*    hA  = (float*)d_ws;
    float*    hB  = hA + (size_t)NN * HH;
    unsigned* agg = (unsigned*)(hB + (size_t)NN * HH);

    const int elemBlocks = (NN * HH) / 256;   // 25000 exactly
    const int edgeBlocks = (EE * 16) / 256;   // 75000 exactly
    const int tileBlocks = NN / 16;           // 6250 exactly

    k_encoder<<<elemBlocks, 256, 0, stream>>>(x, W_enc, b_enc, hA);

    float* hc = hA;
    float* hn = hB;
    for (int l = 0; l < LL; ++l) {
        k_init_agg<<<elemBlocks, 256, 0, stream>>>(agg);
        k_message<<<edgeBlocks, 256, 0, stream>>>(
            hc, ei, ea, W_edge + (size_t)l * HH, b_edge + (size_t)l * HH, agg);
        k_update<<<tileBlocks, 128, 0, stream>>>(
            hc, agg, W_upd + (size_t)l * 2 * HH * HH, b_upd + (size_t)l * HH,
            bn_g + (size_t)l * HH, bn_b + (size_t)l * HH,
            bn_m + (size_t)l * HH, bn_v + (size_t)l * HH, hn);
        float* t = hc; hc = hn; hn = t;
    }

    k_pred1<<<tileBlocks, 128, 0, stream>>>(hc, W_p1, b_p1, hn);
    k_pred2<<<(NN + 255) / 256, 256, 0, stream>>>(hn, W_p2, b_p2, out);
}
